// RecurrentNVP2D_72627896975389
// MI455X (gfx1250) — matrix operand order repaired and checked by reading, not compiled, not run on hardware
//
#include <hip/hip_runtime.h>
#include <hip/hip_bf16.h>

// ---------------------------------------------------------------------------
// RecurrentNVP2D on MI455X (gfx1250): fused 2-layer GRU stacks with bf16 WMMA.
//   B=32, T=2048, D=128; layer1: 128->256, layer2: 256->128, LeakyReLU(0.3).
// One persistent workgroup per stack (16 waves, wave32). State h1/h2 live in
// LDS (bf16); per timestep we do h@U and x@W with v_wmma_f32_16x16x32_bf16.
// Weights are pre-packed in ws into the exact B-fragment lane layout so each
// lane fetches its 16 bf16 elements with a single 32B load (L2-resident).
// ---------------------------------------------------------------------------

typedef __attribute__((ext_vector_type(16))) __bf16 v16bf;
typedef __attribute__((ext_vector_type(8)))  __bf16 v8bf;
typedef __attribute__((ext_vector_type(8)))  float  v8f;

#define B_       32
#define T_       2048
#define D_       128
#define H1_      256
#define H2_      128
#define NTHREADS 512   // 16 wave32 waves
#define ALPHA_   0.3f

// ---------------------------------------------------------------------------

__device__ __forceinline__ v8f wmma_bf16(v16bf a, v16bf b, v8f c) {
  // (neg_a, A, neg_b, B, c_mod, C, reuse_a, reuse_b)
  return __builtin_amdgcn_wmma_f32_16x16x32_bf16(false, a, false, b,
                                                 (short)0, c, false, false);
}

__device__ __forceinline__ float sigmoidf_(float x) {
  return 1.0f / (1.0f + __expf(-x));
}

// A-fragment (16x32 bf16, MxK) from an LDS matrix with row length LD.
// ISA 7.12.2 layout: lane l holds row M = l&15; K runs are
//   kb..kb+7 and kb+16..kb+23 with kb = kt*32 + (l>=16 ? 8 : 0).
template<int LD>
__device__ __forceinline__ v16bf load_afrag(const __bf16* sh, int mtile, int kt) {
  const int l  = threadIdx.x & 31;
  const int m  = mtile * 16 + (l & 15);
  const int kb = kt * 32 + ((l & 16) >> 1);
  const __bf16* p = sh + m * LD + kb;
  v8bf lo = *reinterpret_cast<const v8bf*>(p);       // ds_load_b128
  v8bf hi = *reinterpret_cast<const v8bf*>(p + 16);  // ds_load_b128
  v16bf a;
#pragma unroll
  for (int i = 0; i < 8; ++i) { a[i] = lo[i]; a[8 + i] = hi[i]; }
  return a;
}

// B-fragment (32x16 bf16, KxN) from the pre-packed weight image:
// fragment (kt,nt), lane l -> 16 contiguous bf16 (32 bytes).
__device__ __forceinline__ v16bf load_bfrag(const __bf16* packed, int ntTotal,
                                            int kt, int nt) {
  const int l = threadIdx.x & 31;
  const __bf16* p = packed + (((size_t)kt * ntTotal + nt) * 32 + l) * 16;
  return *reinterpret_cast<const v16bf*>(p);
}

// ---------------------------------------------------------------------------
// One GRU step (rec = h@U + b1, xw = x@W + b0, Keras reset_after gates).
// Each wave owns UPW units; unit = (mtile, ntrip) covering output columns
// n in [ntrip*16, ntrip*16+16) of each of the 3 gate blocks.
// ---------------------------------------------------------------------------
template<int DIN, int H, int UPW>
__device__ __forceinline__ void gru_compute(const __bf16* shx, const __bf16* shh,
                                            const __bf16* Wp, const __bf16* Up,
                                            const float bx[3], const float br[3],
                                            int ntrip, const int* mt,
                                            float hn[][8]) {
  constexpr int KX  = DIN / 32;
  constexpr int KH  = H / 32;
  constexpr int NH  = H / 16;
  constexpr int NT3 = (3 * H) / 16;
  const v8f vzero = {0.f, 0.f, 0.f, 0.f, 0.f, 0.f, 0.f, 0.f};

  v8f accx[UPW][3], accr[UPW][3];
#pragma unroll
  for (int s = 0; s < UPW; ++s)
#pragma unroll
    for (int g = 0; g < 3; ++g) { accx[s][g] = vzero; accr[s][g] = vzero; }

  // ---- x_t @ W ----
#pragma unroll
  for (int kt = 0; kt < KX; ++kt) {
    v16bf a[UPW];
#pragma unroll
    for (int s = 0; s < UPW; ++s) a[s] = load_afrag<DIN>(shx, mt[s], kt);
#pragma unroll
    for (int g = 0; g < 3; ++g) {
      v16bf bw = load_bfrag(Wp, NT3, kt, g * NH + ntrip);
#pragma unroll
      for (int s = 0; s < UPW; ++s) accx[s][g] = wmma_bf16(a[s], bw, accx[s][g]);
    }
  }
  // ---- h @ U ----
#pragma unroll
  for (int kt = 0; kt < KH; ++kt) {
    v16bf a[UPW];
#pragma unroll
    for (int s = 0; s < UPW; ++s) a[s] = load_afrag<H>(shh, mt[s], kt);
#pragma unroll
    for (int g = 0; g < 3; ++g) {
      v16bf bu = load_bfrag(Up, NT3, kt, g * NH + ntrip);
#pragma unroll
      for (int s = 0; s < UPW; ++s) accr[s][g] = wmma_bf16(a[s], bu, accr[s][g]);
    }
  }

  // ---- gates (C/D layout: elem i -> M = mtile*16 + i + (l>=16?8:0), N = l&15) --
  const int l = threadIdx.x & 31;
  const int n = ntrip * 16 + (l & 15);
#pragma unroll
  for (int s = 0; s < UPW; ++s) {
#pragma unroll
    for (int i = 0; i < 8; ++i) {
      const int M = mt[s] * 16 + i + ((l & 16) ? 8 : 0);
      const float z  = sigmoidf_(accx[s][0][i] + bx[0] + accr[s][0][i] + br[0]);
      const float r  = sigmoidf_(accx[s][1][i] + bx[1] + accr[s][1][i] + br[1]);
      const float hh = accx[s][2][i] + bx[2] + r * (accr[s][2][i] + br[2]);
      const float ho = (float)shh[M * H + n];
      hn[s][i] = z * ho + (1.0f - z) * hh;
    }
  }
}

// ---------------------------------------------------------------------------

struct StackArgs {
  const float* x;   long xld;     // input [B,T,*], row stride xld, cols 0..127
  float* y;                       // output logits/tout [B,T,128] (leaky applied)
  const __bf16 *Wp0, *Up0, *Wp1, *Up1;   // packed bf16 fragment images
  const float *b0, *b1;                  // biases (2,3H) f32
};

__global__ __launch_bounds__(NTHREADS)
void gru_stack_kernel(StackArgs a0, StackArgs a1) {
  __shared__ __bf16 shx [B_ * D_];   //  8 KB staged x_t
  __shared__ __bf16 shh1[B_ * H1_];  // 16 KB layer-1 state
  __shared__ __bf16 shy1[B_ * H1_];  // 16 KB leaky(layer-1 out) -> layer-2 in
  __shared__ __bf16 shh2[B_ * H2_];  //  8 KB layer-2 state

  const StackArgs a = (blockIdx.x == 0) ? a0 : a1;
  const int tid = threadIdx.x;
  const int wid = tid >> 5;
  const int l   = tid & 31;

  for (int i = tid; i < B_ * H1_; i += NTHREADS) shh1[i] = (__bf16)0.0f;
  for (int i = tid; i < B_ * H2_; i += NTHREADS) shh2[i] = (__bf16)0.0f;

  // Layer 1: H=256 -> 16 N-triplets x 2 M-tiles = 32 units, 2 per wave.
  const int ntrip1 = wid;
  const int mt1[2] = {0, 1};
  // Layer 2: H=128 -> 8 N-triplets x 2 M-tiles = 16 units, 1 per wave.
  const int ntrip2 = wid & 7;
  const int mt2[1] = { wid >> 3 };

  const int n1 = ntrip1 * 16 + (l & 15);
  const int n2 = ntrip2 * 16 + (l & 15);
  float bx1[3], br1[3], bx2[3], br2[3];
#pragma unroll
  for (int g = 0; g < 3; ++g) {
    bx1[g] = a.b0[g * H1_ + n1];
    br1[g] = a.b0[3 * H1_ + g * H1_ + n1];
    bx2[g] = a.b1[g * H2_ + n2];
    br2[g] = a.b1[3 * H2_ + g * H2_ + n2];
  }
  __syncthreads();

  for (int t = 0; t < T_; ++t) {
    // stage x_t (f32 global -> bf16 LDS), prefetch next step
    for (int i = tid; i < B_ * D_; i += NTHREADS) {
      const int m = i >> 7, k = i & (D_ - 1);
      shx[i] = (__bf16)a.x[((size_t)m * T_ + t) * a.xld + k];
      if (t + 1 < T_)
        __builtin_prefetch(&a.x[((size_t)m * T_ + t + 1) * a.xld + k], 0, 1);
    }
    __syncthreads();

    float hn1[2][8];
    gru_compute<D_, H1_, 2>(shx, shh1, a.Wp0, a.Up0, bx1, br1, ntrip1, mt1, hn1);
    __syncthreads();
#pragma unroll
    for (int s = 0; s < 2; ++s)
#pragma unroll
      for (int i = 0; i < 8; ++i) {
        const int M = mt1[s] * 16 + i + ((l & 16) ? 8 : 0);
        const float h = hn1[s][i];
        shh1[M * H1_ + n1] = (__bf16)h;
        const float ly = (h >= 0.0f) ? h : ALPHA_ * h;
        shy1[M * H1_ + n1] = (__bf16)ly;
      }
    __syncthreads();

    float hn2[1][8];
    gru_compute<H1_, H2_, 1>(shy1, shh2, a.Wp1, a.Up1, bx2, br2, ntrip2, mt2, hn2);
    __syncthreads();
#pragma unroll
    for (int i = 0; i < 8; ++i) {
      const int M = mt2[0] * 16 + i + ((l & 16) ? 8 : 0);
      const float h = hn2[0][i];
      shh2[M * H2_ + n2] = (__bf16)h;
      const float ly = (h >= 0.0f) ? h : ALPHA_ * h;
      a.y[((size_t)M * T_ + t) * H2_ + n2] = ly;
    }
    __syncthreads();
  }
}

// ---------------------------------------------------------------------------
// Pack a f32 [K,N] weight into per-lane B-fragment layout, bf16:
//   dst[((kt*Nt + nt)*32 + lane)*16 + e] = src[kt*32 + (lane>=16?8:0) + e + (e>=8?8:0)][nt*16 + (lane&15)]
// ---------------------------------------------------------------------------
__global__ void pack_bfrag_kernel(const float* __restrict__ src,
                                  __bf16* __restrict__ dst, int K, int N) {
  const int tid = blockIdx.x * blockDim.x + threadIdx.x;
  const int ntTotal = N / 16;
  const int total = (K / 32) * ntTotal * 32;
  if (tid >= total) return;
  const int l = tid & 31;
  const int frag = tid >> 5;
  const int nt = frag % ntTotal, kt = frag / ntTotal;
  const int n = nt * 16 + (l & 15);
  const int kbase = kt * 32 + ((l & 16) ? 8 : 0);
  __bf16* out = dst + (size_t)tid * 16;
#pragma unroll
  for (int e = 0; e < 16; ++e)
    out[e] = (__bf16)src[(size_t)(kbase + e + (e >= 8 ? 8 : 0)) * N + n];
}

// v = u * exp(logit) + tout  ->  out[:, :, colOff:colOff+128]
__global__ void combine_kernel(const float* __restrict__ u,
                               const float* __restrict__ logit,
                               const float* __restrict__ tout,
                               float* __restrict__ out, int colOff) {
  const size_t i = (size_t)blockIdx.x * blockDim.x + threadIdx.x;
  if (i >= (size_t)B_ * T_ * H2_) return;
  const size_t row = i >> 7;
  const int k = (int)(i & 127);
  out[row * (2 * D_) + colOff + k] = u[i] * __expf(logit[i]) + tout[i];
}

// dst[r] = sum_k logit[r,k] (+ addin[r])
__global__ void rowsum_kernel(const float* __restrict__ logit,
                              const float* __restrict__ addin,
                              float* __restrict__ dst) {
  const int r = blockIdx.x * blockDim.x + threadIdx.x;
  if (r >= B_ * T_) return;
  const float* p = logit + (size_t)r * H2_;
  float s = 0.0f;
#pragma unroll 4
  for (int k = 0; k < H2_; ++k) s += p[k];
  dst[r] = addin ? (s + addin[r]) : s;
}

// ---------------------------------------------------------------------------

extern "C" void kernel_launch(void* const* d_in, const int* in_sizes, int n_in,
                              void* d_out, int out_size, void* d_ws, size_t ws_size,
                              hipStream_t stream) {
  if (n_in < 26) return;
  (void)out_size; (void)ws_size;

  // Input layout: insertion order [u1,u2, s1(6), t1(6), s2(6), t2(6)]
  // or pytree-sorted [s1,s2,t1,t2 (6 each), u1, u2].
  const bool uFirst = (in_sizes[0] == B_ * T_ * D_);
  const int pbase = uFirst ? 2 : 0;
  const int ubase = uFirst ? 0 : 24;
  int phys[4];                     // logical {S1,T1,S2,T2} -> physical group
  if (uFirst) { phys[0]=0; phys[1]=1; phys[2]=2; phys[3]=3; }
  else        { phys[0]=0; phys[1]=2; phys[2]=1; phys[3]=3; }

  const float* u1 = (const float*)d_in[ubase + 0];
  const float* u2 = (const float*)d_in[ubase + 1];
  auto P = [&](int ls, int j) -> const float* {
    return (const float*)d_in[pbase + phys[ls] * 6 + j];
  };

  // workspace carve-up
  const size_t szW0 = (size_t)D_  * 3 * H1_;   //  98304
  const size_t szU0 = (size_t)H1_ * 3 * H1_;   // 196608
  const size_t szW1 = (size_t)H1_ * 3 * H2_;   //  98304
  const size_t szU1 = (size_t)H2_ * 3 * H2_;   //  49152
  const size_t per  = szW0 + szU0 + szW1 + szU1;
  __bf16* packed = (__bf16*)d_ws;
  float* fbuf    = (float*)((char*)d_ws + 4 * per * sizeof(__bf16));
  float* logitB  = fbuf;                                // [B*T*128]
  float* toutB   = logitB + (size_t)B_ * T_ * H2_;      // [B*T*128]
  float* loss1   = toutB  + (size_t)B_ * T_ * H2_;      // [B*T]
  float* out     = (float*)d_out;

  // pack all weights into bf16 fragment images (L2-resident, ~3.5 MB)
  const int Kdim[4] = {D_, H1_, H1_, H2_};
  const int Ndim[4] = {3*H1_, 3*H1_, 3*H2_, 3*H2_};
  const int srcIdx[4] = {0, 1, 3, 4};
  for (int ls = 0; ls < 4; ++ls) {
    __bf16* base = packed + (size_t)ls * per;
    size_t off = 0;
    for (int m = 0; m < 4; ++m) {
      const int total = (Kdim[m] / 32) * (Ndim[m] / 16) * 32;
      pack_bfrag_kernel<<<(total + 255) / 256, 256, 0, stream>>>(
          P(ls, srcIdx[m]), base + off, Kdim[m], Ndim[m]);
      off += (size_t)Kdim[m] * Ndim[m];
    }
  }

  auto mk = [&](int ls, const float* x, long xld, float* y) -> StackArgs {
    StackArgs s;
    __bf16* base = packed + (size_t)ls * per;
    s.x = x; s.xld = xld; s.y = y;
    s.Wp0 = base;
    s.Up0 = base + szW0;
    s.Wp1 = base + szW0 + szU0;
    s.Up1 = base + szW0 + szU0 + szW1;
    s.b0 = P(ls, 2); s.b1 = P(ls, 5);
    return s;
  };

  const size_t nElem = (size_t)B_ * T_ * H2_;
  const int cBlocks = (int)((nElem + 255) / 256);
  const int rBlocks = (B_ * T_ + 255) / 256;

  // Stage A: logit1 = stack(u2, s1); t1out = stack(u2, t1)
  gru_stack_kernel<<<2, NTHREADS, 0, stream>>>(mk(0, u2, D_, logitB),
                                               mk(1, u2, D_, toutB));
  // v1 = u1*exp(logit1) + t1out -> out[:, :, 0:128];  loss1 = sum(logit1)
  combine_kernel<<<cBlocks, 256, 0, stream>>>(u1, logitB, toutB, out, 0);
  rowsum_kernel<<<rBlocks, 256, 0, stream>>>(logitB, nullptr, loss1);

  // Stage B: read v1 straight out of d_out (row stride 256, cols 0..127)
  gru_stack_kernel<<<2, NTHREADS, 0, stream>>>(mk(2, out, 2 * D_, logitB),
                                               mk(3, out, 2 * D_, toutB));
  // v2 = u2*exp(logit2) + t2out -> out[:, :, 128:256]; loss = loss1 + sum(logit2)
  combine_kernel<<<cBlocks, 256, 0, stream>>>(u2, logitB, toutB, out, D_);
  rowsum_kernel<<<rBlocks, 256, 0, stream>>>(logitB, loss1,
                                             out + (size_t)B_ * T_ * 2 * D_);
}
